// RNN_69114613728639
// MI455X (gfx1250) — matrix operand order repaired and checked by reading, not compiled, not run on hardware
//
#include <hip/hip_runtime.h>
#include <hip/hip_bf16.h>

// ---------------------------------------------------------------------------
// LSTM text classifier for MI455X (gfx1250, wave32, WMMA).
//   prep_kernel : pack [W_hh | W_ih] -> bf16 Wcat[1024][512], fc1_w -> bf16,
//                 bias = b_ih + b_hh
//   lstm_kernel : 4 persistent workgroups (16 batch rows each), 1024 steps of
//                 gates[16,1024] = [h | x_e][16,512] @ Wcat^T  via
//                 v_wmma_f32_16x16x32_bf16, cell update in LDS.
//   head_kernel : fc1 (WMMA) + training-mode BatchNorm + fc2.
// ---------------------------------------------------------------------------

#define VOCAB   50000
#define EMBED   256
#define HIDDEN  256
#define GATES   1024          // 4*HIDDEN
#define KDIM    512           // HIDDEN + EMBED (fused recurrent+input GEMM)
#define BATCH   64
#define SEQ     1024
#define BCHUNK  16            // batch rows per workgroup (== WMMA M)
#define NCHUNKS 4
#define FC      256
#define NOUT    10
#define BN_EPS  1e-5f

typedef __bf16 bf16;
typedef __attribute__((ext_vector_type(16))) __bf16 bf16x16;
typedef __attribute__((ext_vector_type(8)))  float  f32x8;

union Frag { bf16x16 v; uint4 q[2]; };

__device__ __forceinline__ float fsig(float x) {
    return 1.0f / (1.0f + __expf(-x));
}
__device__ __forceinline__ float ftanh(float x) {
    x = fminf(fmaxf(x, -15.0f), 15.0f);
    float e = __expf(2.0f * x);
    return (e - 1.0f) / (e + 1.0f);
}

// ---------------------------------------------------------------------------
// Kernel 0: weight packing.
// Wcat[n][k] = (k < 256) ? W_hh[n][k] : W_ih[n][k-256]   (bf16, row-major 512)
// fc1wb[n][k] = fc1_w[n][k] (bf16), biasg[j] = b_ih[j] + b_hh[j]
// ---------------------------------------------------------------------------
__global__ void prep_kernel(const float* __restrict__ W_ih,
                            const float* __restrict__ W_hh,
                            const float* __restrict__ b_ih,
                            const float* __restrict__ b_hh,
                            const float* __restrict__ fc1_w,
                            bf16* __restrict__ Wcat,
                            bf16* __restrict__ fc1wb,
                            float* __restrict__ biasg) {
    int i = blockIdx.x * blockDim.x + threadIdx.x;
    if (i < GATES * KDIM) {
        int n = i >> 9, k = i & (KDIM - 1);
        float v = (k < HIDDEN) ? W_hh[n * HIDDEN + k]
                               : W_ih[n * EMBED + (k - HIDDEN)];
        Wcat[i] = (bf16)v;
    }
    if (i < FC * HIDDEN) fc1wb[i] = (bf16)fc1_w[i];
    if (i < GATES)       biasg[i] = b_ih[i] + b_hh[i];
}

// ---------------------------------------------------------------------------
// Kernel 1: persistent LSTM recurrence. grid = 4 WGs x 256 threads (8 waves).
// LDS: A-panel [16][512] bf16 (h | x_e), gates [16][1024] f32, c [16][256] f32.
// ---------------------------------------------------------------------------
__global__ __launch_bounds__(256)
void lstm_kernel(const int* __restrict__ x,
                 const float* __restrict__ emb,
                 const bf16* __restrict__ Wcat,
                 const float* __restrict__ biasg,
                 float* __restrict__ hlast) {
    __shared__ bf16  Abuf[BCHUNK * KDIM];      // 16 KB: cols [0,256)=h, [256,512)=x_e
    __shared__ float gatesb[BCHUNK * GATES];   // 64 KB
    __shared__ float cbuf[BCHUNK * HIDDEN];    // 16 KB
    __shared__ float biasb[GATES];             //  4 KB

    const int tid   = threadIdx.x;
    const int chunk = blockIdx.x;
    const int lane  = tid & 31;
    const int wv    = tid >> 5;         // wave id 0..7
    const int laneN = lane & 15;        // N (or M) within tile
    const int half  = lane >> 4;        // lane half selects K sub-block
    const int b     = tid >> 4;         // batch row 0..15 (elementwise phases)
    const int seg   = tid & 15;         // 16-wide column segment

    // ---- init state / bias
#pragma unroll
    for (int r = 0; r < 4; ++r) biasb[tid * 4 + r] = biasg[tid * 4 + r];
#pragma unroll
    for (int jj = 0; jj < 16; ++jj) {
        cbuf[b * HIDDEN + seg * 16 + jj] = 0.0f;
        Abuf[b * KDIM + seg * 16 + jj]   = (bf16)0.0f;
    }
    __syncthreads();

    for (int t = 0; t < SEQ; ++t) {
        // ---- gather embedded token for this step into A-panel cols [256,512)
        {
            int idx = x[(chunk * BCHUNK + b) * SEQ + t];
            const float4* ep = (const float4*)&emb[(long)idx * EMBED + seg * 16];
#pragma unroll
            for (int q = 0; q < 4; ++q) {
                float4 v = ep[q];
                bf16* dp = &Abuf[b * KDIM + HIDDEN + seg * 16 + q * 4];
                dp[0] = (bf16)v.x; dp[1] = (bf16)v.y;
                dp[2] = (bf16)v.z; dp[3] = (bf16)v.w;
            }
        }
        __syncthreads();

        // ---- gates[16,1024] = A[16,512] @ Wcat^T   (bf16 WMMA, f32 acc)
        // wave wv owns N-tiles wv*8 .. wv*8+7
        f32x8 acc[8];
#pragma unroll
        for (int nt = 0; nt < 8; ++nt)
#pragma unroll
            for (int r = 0; r < 8; ++r) acc[nt][r] = 0.0f;

        for (int kb = 0; kb < 16; ++kb) {
            // A fragment (16x32 bf16): two 16B chunks per lane from LDS
            Frag af;
            const bf16* ap = &Abuf[laneN * KDIM + kb * 32 + half * 8];
            af.q[0] = *(const uint4*)ap;
            af.q[1] = *(const uint4*)(ap + 16);
#pragma unroll
            for (int nt = 0; nt < 8; ++nt) {
                // B fragment (32x16 bf16): two 16B chunks per lane from weight row
                Frag bfr;
                const bf16* bp = &Wcat[((wv * 8 + nt) * 16 + laneN) * KDIM
                                       + kb * 32 + half * 8];
                bfr.q[0] = *(const uint4*)bp;
                bfr.q[1] = *(const uint4*)(bp + 16);
                acc[nt] = __builtin_amdgcn_wmma_f32_16x16x32_bf16(
                    false, af.v, false, bfr.v, (short)0, acc[nt], false, false);
            }
        }

        // ---- spill gate accumulators to LDS (row = r + 8*half, col = lane%16)
#pragma unroll
        for (int nt = 0; nt < 8; ++nt) {
            int nb = (wv * 8 + nt) * 16 + laneN;
#pragma unroll
            for (int r = 0; r < 8; ++r)
                gatesb[(r + 8 * half) * GATES + nb] = acc[nt][r];
        }
        __syncthreads();

        // ---- LSTM cell update: thread owns (row b, cols seg*16..+15)
#pragma unroll
        for (int jj = 0; jj < 16; ++jj) {
            int j = seg * 16 + jj;
            float gi = gatesb[b * GATES + j             ] + biasb[j];
            float gf = gatesb[b * GATES + HIDDEN + j    ] + biasb[HIDDEN + j];
            float gg = gatesb[b * GATES + 2 * HIDDEN + j] + biasb[2 * HIDDEN + j];
            float go = gatesb[b * GATES + 3 * HIDDEN + j] + biasb[3 * HIDDEN + j];
            float c = fsig(gf) * cbuf[b * HIDDEN + j] + fsig(gi) * ftanh(gg);
            float h = fsig(go) * ftanh(c);
            cbuf[b * HIDDEN + j] = c;
            Abuf[b * KDIM + j]   = (bf16)h;     // feeds next step's A-panel
            if (t == SEQ - 1)
                hlast[(chunk * BCHUNK + b) * HIDDEN + j] = h;
        }
        __syncthreads();
    }
}

// ---------------------------------------------------------------------------
// Kernel 2: fc1 (WMMA) + BatchNorm (batch statistics) + fc2. One workgroup.
// ---------------------------------------------------------------------------
__global__ __launch_bounds__(256)
void head_kernel(const float* __restrict__ hlast,
                 const bf16* __restrict__ fc1wb,
                 const float* __restrict__ fc1_b,
                 const float* __restrict__ gamma,
                 const float* __restrict__ beta,
                 const float* __restrict__ fc2_w,
                 const float* __restrict__ fc2_b,
                 float* __restrict__ out) {
    __shared__ bf16  hbuf[BATCH * HIDDEN];   // 32 KB
    __shared__ float fc1buf[BATCH * FC];     // 64 KB

    const int tid   = threadIdx.x;
    const int lane  = tid & 31;
    const int wv    = tid >> 5;
    const int laneN = lane & 15;
    const int half  = lane >> 4;

    for (int i = tid; i < BATCH * HIDDEN; i += 256)
        hbuf[i] = (bf16)hlast[i];
    __syncthreads();

    // fc1 = h @ fc1_w^T + b : 4 M-tiles x 16 N-tiles; wave wv -> N-tiles {2wv, 2wv+1}
#pragma unroll
    for (int tn = 0; tn < 2; ++tn) {
        int ntile = wv * 2 + tn;
#pragma unroll
        for (int mt = 0; mt < 4; ++mt) {
            f32x8 acc;
#pragma unroll
            for (int r = 0; r < 8; ++r) acc[r] = 0.0f;
#pragma unroll
            for (int kb = 0; kb < 8; ++kb) {
                Frag af, bfr;
                const bf16* ap = &hbuf[(mt * 16 + laneN) * HIDDEN + kb * 32 + half * 8];
                af.q[0] = *(const uint4*)ap;
                af.q[1] = *(const uint4*)(ap + 16);
                const bf16* bp = &fc1wb[(ntile * 16 + laneN) * HIDDEN + kb * 32 + half * 8];
                bfr.q[0] = *(const uint4*)bp;
                bfr.q[1] = *(const uint4*)(bp + 16);
                acc = __builtin_amdgcn_wmma_f32_16x16x32_bf16(
                    false, af.v, false, bfr.v, (short)0, acc, false, false);
            }
            int col = ntile * 16 + laneN;
#pragma unroll
            for (int r = 0; r < 8; ++r)
                fc1buf[(mt * 16 + r + 8 * half) * FC + col] = acc[r] + fc1_b[col];
        }
    }
    __syncthreads();

    // BatchNorm1d, training-mode batch stats (biased variance). One column/thread;
    // per-instruction LDS access across threads is contiguous -> conflict-free.
    {
        int j = tid;
        float s = 0.0f, ss = 0.0f;
        for (int bb = 0; bb < BATCH; ++bb) {
            float v = fc1buf[bb * FC + j];
            s += v; ss += v * v;
        }
        float mu    = s * (1.0f / BATCH);
        float var   = ss * (1.0f / BATCH) - mu * mu;
        float scale = rsqrtf(var + BN_EPS) * gamma[j];
        float shift = beta[j] - mu * scale;
        for (int bb = 0; bb < BATCH; ++bb)
            fc1buf[bb * FC + j] = fc1buf[bb * FC + j] * scale + shift;
    }
    __syncthreads();

    // fc2: [64,10] plain VALU dot products (163K MACs, negligible)
    for (int idx = tid; idx < BATCH * NOUT; idx += 256) {
        int bb = idx / NOUT, o = idx % NOUT;
        float s = fc2_b[o];
        for (int j = 0; j < FC; ++j)
            s += fc1buf[bb * FC + j] * fc2_w[o * FC + j];
        out[idx] = s;
    }
}

// ---------------------------------------------------------------------------
extern "C" void kernel_launch(void* const* d_in, const int* in_sizes, int n_in,
                              void* d_out, int out_size, void* d_ws, size_t ws_size,
                              hipStream_t stream) {
    (void)in_sizes; (void)n_in; (void)out_size; (void)ws_size;
    const int*   x     = (const int*)d_in[0];
    const float* emb   = (const float*)d_in[1];
    const float* W_ih  = (const float*)d_in[2];
    const float* W_hh  = (const float*)d_in[3];
    const float* b_ih  = (const float*)d_in[4];
    const float* b_hh  = (const float*)d_in[5];
    const float* fc1_w = (const float*)d_in[6];
    const float* fc1_b = (const float*)d_in[7];
    const float* gamma = (const float*)d_in[8];
    const float* beta  = (const float*)d_in[9];
    const float* fc2_w = (const float*)d_in[10];
    const float* fc2_b = (const float*)d_in[11];
    float* out = (float*)d_out;

    char* ws = (char*)d_ws;
    bf16*  Wcat  = (bf16*)(ws);                                    // 1 MB
    bf16*  fc1wb = (bf16*)(ws + (1 << 20));                        // 128 KB
    float* biasg = (float*)(ws + (1 << 20) + (1 << 17));           // 4 KB
    float* hlast = (float*)(ws + (1 << 20) + (1 << 17) + (1 << 12)); // 64 KB

    prep_kernel<<<(GATES * KDIM + 255) / 256, 256, 0, stream>>>(
        W_ih, W_hh, b_ih, b_hh, fc1_w, Wcat, fc1wb, biasg);
    lstm_kernel<<<NCHUNKS, 256, 0, stream>>>(x, emb, Wcat, biasg, hlast);
    head_kernel<<<1, 256, 0, stream>>>(hlast, fc1wb, fc1_b, gamma, beta,
                                       fc2_w, fc2_b, out);
}
